// IPMPDenoiser_33767032881716
// MI455X (gfx1250) — matrix operand order repaired and checked by reading, not compiled, not run on hardware
//
#include <hip/hip_runtime.h>
#include <hip/hip_bf16.h>
#include <math.h>

#define NRES 4096
#define KNN  30
#define NEDGE (NRES*KNN)

typedef _Float16 f16;
typedef __attribute__((ext_vector_type(16))) _Float16 v16h;
typedef __attribute__((ext_vector_type(8)))  _Float16 v8h;
typedef __attribute__((ext_vector_type(8)))  float    v8f;

// ------------------------------ small helpers ------------------------------
__device__ __forceinline__ float3 f3ld(const float* p){ return make_float3(p[0],p[1],p[2]); }
__device__ __forceinline__ float3 f3sub(float3 a,float3 b){ return make_float3(a.x-b.x,a.y-b.y,a.z-b.z); }
__device__ __forceinline__ float3 f3cross(float3 a,float3 b){
  return make_float3(a.y*b.z-a.z*b.y, a.z*b.x-a.x*b.z, a.x*b.y-a.y*b.x); }
__device__ __forceinline__ float  f3dot(float3 a,float3 b){ return a.x*b.x+a.y*b.y+a.z*b.z; }
__device__ __forceinline__ float3 f3unit(float3 a,float eps){
  float n = sqrtf(f3dot(a,a)) + eps; return make_float3(a.x/n,a.y/n,a.z/n); }

// ------------------------------ WMMA GEMM ------------------------------
// C[M,N] = act(A[M,K] @ B[K,N] + bias) ; A,B f16 row-major, f32 accumulate.
// block = 128 thr = 4 waves; block tile 64(M) x 64(N); wave w -> rows 16w..16w+15.
// K multiple of 32, M multiple of 64, N multiple of 64 (everything pre-padded).
// Double-buffered LDS: next tile's global loads are issued before this tile's
// WMMAs; one barrier per K-step.
__global__ __launch_bounds__(128)
void wmma_gemm_kernel(const f16* __restrict__ A, const f16* __restrict__ B,
                      const float* __restrict__ bias,
                      float* __restrict__ Cf, f16* __restrict__ Ch,
                      const float* __restrict__ rowmask,
                      int M, int N, int K, int relu)
{
  __shared__ __attribute__((aligned(64))) f16 lA[2][64*32];
  __shared__ __attribute__((aligned(64))) f16 lB[2][32*64];
  const int tid  = threadIdx.x;
  const int wave = tid >> 5;
  const int lane = tid & 31;
  const int m0 = blockIdx.x * 64;
  const int n0 = blockIdx.y * 64;
  const int g  = lane >> 4;      // half-lane group
  const int ml = lane & 15;

  v8f acc[4];
  const v8f vzero = {0.f,0.f,0.f,0.f,0.f,0.f,0.f,0.f};
  #pragma unroll
  for (int c = 0; c < 4; ++c) acc[c] = vzero;

  // cooperative staging coordinates
  const int ar = tid >> 1;            // A row 0..63
  const int ah = (tid & 1) * 16;      // A col half 0 / 16
  const int br = tid >> 2;            // B row 0..31
  const int bq = (tid & 3) * 16;      // B col quarter

  const f16* gA = A + (size_t)(m0 + ar) * K + ah;     // + it*32
  const f16* gB = B + (size_t)br * N + n0 + bq;       // + it*32*N

  // prologue: stage tile 0
  {
    v8h a0 = *(const v8h*)(gA);
    v8h a1 = *(const v8h*)(gA + 8);
    v8h b0 = *(const v8h*)(gB);
    v8h b1 = *(const v8h*)(gB + 8);
    *(v8h*)&lA[0][ar*32 + ah]     = a0;
    *(v8h*)&lA[0][ar*32 + ah + 8] = a1;
    *(v8h*)&lB[0][br*64 + bq]     = b0;
    *(v8h*)&lB[0][br*64 + bq + 8] = b1;
  }
  __syncthreads();

  const int nsteps = K >> 5;
  int cur = 0;
  for (int it = 0; it < nsteps; ++it) {
    const bool hasNext = (it + 1) < nsteps;
    v8h na0{}, na1{}, nb0{}, nb1{};
    if (hasNext) {
      const f16* pA = gA + (it + 1) * 32;
      const f16* pB = gB + (size_t)(it + 1) * 32 * N;
      na0 = *(const v8h*)(pA);
      na1 = *(const v8h*)(pA + 8);
      nb0 = *(const v8h*)(pB);
      nb1 = *(const v8h*)(pB + 8);
      if (it + 2 < nsteps) __builtin_prefetch(gA + (it + 2) * 32, 0, 0);
    }

    // A fragment: 16-bit A 16x32 layout (ISA 7.12.2):
    // elems 0..7 -> K = g*8+e ; elems 8..15 -> K = 16+g*8+e
    const f16* arow = &lA[cur][(wave*16 + ml) * 32];
    v8h alo = *(const v8h*)&arow[g*8];
    v8h ahi = *(const v8h*)&arow[16 + g*8];
    v16h afrag = __builtin_shufflevector(alo, ahi, 0,1,2,3,4,5,6,7,8,9,10,11,12,13,14,15);

    // B fragment: lane%16 = N col, elems follow A's K order
    const f16* brow = &lB[cur][ml];
    #pragma unroll
    for (int c = 0; c < 4; ++c) {
      v16h bfrag;
      #pragma unroll
      for (int s = 0; s < 16; ++s)
        bfrag[s] = brow[((s < 8) ? (g*8 + s) : (16 + g*8 + (s - 8))) * 64 + c*16];
      acc[c] = __builtin_amdgcn_wmma_f32_16x16x32_f16(
                 false, afrag, false, bfrag, (short)0, acc[c], false, false);
    }

    if (hasNext) {
      const int nxt = cur ^ 1;
      *(v8h*)&lA[nxt][ar*32 + ah]     = na0;
      *(v8h*)&lA[nxt][ar*32 + ah + 8] = na1;
      *(v8h*)&lB[nxt][br*64 + bq]     = nb0;
      *(v8h*)&lB[nxt][br*64 + bq + 8] = nb1;
      __syncthreads();
      cur = nxt;
    }
  }

  // epilogue: C VGPR r -> row = r (+8 for upper half-lanes), col = lane%16
  #pragma unroll
  for (int c = 0; c < 4; ++c) {
    const int col = n0 + c*16 + ml;
    const float bv = bias ? bias[col] : 0.f;
    #pragma unroll
    for (int r = 0; r < 8; ++r) {
      const int row = m0 + wave*16 + g*8 + r;
      float v = acc[c][r] + bv;
      if (relu) v = v > 0.f ? v : 0.f;
      if (rowmask) v *= rowmask[row];
      const size_t o = (size_t)row * N + col;
      if (Ch) Ch[o] = (f16)v; else Cf[o] = v;
    }
  }
}

// ------------------------------ LayerNorm (cols = 128) ------------------------------
__global__ __launch_bounds__(128)
void ln128_kernel(float* __restrict__ out, const float* __restrict__ in,
                  const float* __restrict__ res, const float* __restrict__ gamma,
                  const float* __restrict__ beta, const float* __restrict__ rowmask,
                  int rows)
{
  const int wave = threadIdx.x >> 5, lane = threadIdx.x & 31;
  const int row = blockIdx.x * 4 + wave;
  if (row >= rows) return;
  const float* ip = in + (size_t)row * 128;
  const float* rp = res ? res + (size_t)row * 128 : nullptr;
  float v[4];
  #pragma unroll
  for (int j = 0; j < 4; ++j) {
    const int c = lane + 32*j;
    v[j] = ip[c] + (rp ? rp[c] : 0.f);
  }
  float s = v[0]+v[1]+v[2]+v[3];
  #pragma unroll
  for (int off = 16; off; off >>= 1) s += __shfl_xor(s, off, 32);
  const float mean = s * (1.f/128.f);
  float q = 0.f;
  #pragma unroll
  for (int j = 0; j < 4; ++j) { const float d = v[j]-mean; q += d*d; }
  #pragma unroll
  for (int off = 16; off; off >>= 1) q += __shfl_xor(q, off, 32);
  const float rstd = rsqrtf(q * (1.f/128.f) + 1e-5f);
  const float m = rowmask ? rowmask[row] : 1.f;
  #pragma unroll
  for (int j = 0; j < 4; ++j) {
    const int c = lane + 32*j;
    out[(size_t)row*128 + c] = ((v[j]-mean)*rstd*gamma[c] + beta[c]) * m;
  }
}

// ------------------------------ geometry kernels ------------------------------
__global__ void dihedral_kernel(const float* __restrict__ bb,
                                const unsigned char* __restrict__ mask,
                                float* __restrict__ dih)
{
  const int i = blockIdx.x * blockDim.x + threadIdx.x;
  if (i >= NRES * 3) return;
  const int n = i / 3, j = i % 3;
  const int t = i - 1;
  float D = 0.f;
  if (t >= 0 && t < 3*NRES - 3) {
    auto X = [&](int k)->float3 {
      const int r = k / 3, a = k % 3;
      return f3ld(bb + (size_t)r*12 + a*3);
    };
    const float eps = 1e-7f;
    float3 x0=X(t), x1=X(t+1), x2=X(t+2), x3=X(t+3);
    float3 u2 = f3unit(f3sub(x1,x0), eps);
    float3 u1 = f3unit(f3sub(x2,x1), eps);
    float3 u0 = f3unit(f3sub(x3,x2), eps);
    float3 n2 = f3unit(f3cross(u2,u1), eps);
    float3 n1 = f3unit(f3cross(u1,u0), eps);
    float cosD = f3dot(n2,n1);
    cosD = fminf(fmaxf(cosD, -1.f+eps), 1.f-eps);
    const float sg = f3dot(u2,n1) >= 0.f ? 1.f : -1.f;
    D = sg * acosf(cosD);
  }
  const float mf = mask[n] ? 1.f : 0.f;
  dih[n*6 + j]     = cosf(D) * mf;
  dih[n*6 + 3 + j] = sinf(D) * mf;
}

__global__ void bb5_kernel(const float* __restrict__ bb, float* __restrict__ bb5)
{
  const int n = blockIdx.x * blockDim.x + threadIdx.x;
  if (n >= NRES) return;
  const float* p = bb + (size_t)n*12;
  float* o = bb5 + (size_t)n*15;
  for (int k = 0; k < 12; ++k) o[k] = p[k];
  float3 N  = f3ld(p+0), CA = f3ld(p+3), C = f3ld(p+6);
  float3 b = f3sub(CA,N), cc = f3sub(C,CA);
  float3 a = f3cross(b,cc);
  o[12] = -0.58273431f*a.x + 0.56802827f*b.x - 0.54067466f*cc.x + CA.x;
  o[13] = -0.58273431f*a.y + 0.56802827f*b.y - 0.54067466f*cc.y + CA.y;
  o[14] = -0.58273431f*a.z + 0.56802827f*b.z - 0.54067466f*cc.z + CA.z;
}

__global__ void quat_kernel(const float* __restrict__ rigids, float* __restrict__ R)
{
  const int n = blockIdx.x * blockDim.x + threadIdx.x;
  if (n >= NRES) return;
  const float* q = rigids + (size_t)n*7;
  const float inv = rsqrtf(q[0]*q[0]+q[1]*q[1]+q[2]*q[2]+q[3]*q[3]);
  const float w=q[0]*inv, x=q[1]*inv, y=q[2]*inv, z=q[3]*inv;
  float* r = R + (size_t)n*9;
  r[0]=1.f-2.f*(y*y+z*z); r[1]=2.f*(x*y-w*z);     r[2]=2.f*(x*z+w*y);
  r[3]=2.f*(x*y+w*z);     r[4]=1.f-2.f*(x*x+z*z); r[5]=2.f*(y*z-w*x);
  r[6]=2.f*(x*z-w*y);     r[7]=2.f*(y*z+w*x);     r[8]=1.f-2.f*(x*x+y*y);
}

__global__ void nodemask_kernel(const unsigned char* __restrict__ mask, float* __restrict__ mf)
{
  const int n = blockIdx.x * blockDim.x + threadIdx.x;
  if (n < NRES) mf[n] = mask[n] ? 1.f : 0.f;
}

__global__ void cvt_a_kernel(const float* __restrict__ in, f16* __restrict__ out, int n)
{
  const int i = blockIdx.x * blockDim.x + threadIdx.x;
  if (i < n) out[i] = (f16)in[i];
}

// ------------------------------ KNN ------------------------------
__global__ __launch_bounds__(128)
void knn_kernel(const float* __restrict__ x, const unsigned char* __restrict__ mask,
                int* __restrict__ srcIdx, float* __restrict__ emask)
{
  __shared__ float d2[512];
  __shared__ float rv[128];
  __shared__ int   ri[128];
  const int n  = blockIdx.x;
  const int g0 = (n >> 9) << 9;                 // graph base (512 per graph)
  const float3 xn = f3ld(x + (size_t)n*3);
  const int tid = threadIdx.x;
  for (int j = tid; j < 512; j += 128) {
    const int cand = g0 + j;
    const float3 xc = f3ld(x + (size_t)cand*3);
    const float dx = xn.x-xc.x, dy = xn.y-xc.y, dz = xn.z-xc.z;
    const bool bad = (cand == n) || !mask[cand] || !mask[n];
    d2[j] = bad ? 1e9f : (dx*dx + dy*dy + dz*dz);
  }
  __syncthreads();
  for (int kk = 0; kk < KNN; ++kk) {
    float best = 1e30f; int bi = 0;
    for (int j = tid; j < 512; j += 128)
      if (d2[j] < best) { best = d2[j]; bi = j; }
    rv[tid] = best; ri[tid] = bi;
    __syncthreads();
    for (int s = 64; s; s >>= 1) {
      if (tid < s && rv[tid + s] < rv[tid]) { rv[tid] = rv[tid+s]; ri[tid] = ri[tid+s]; }
      __syncthreads();
    }
    if (tid == 0) {
      const int bj = ri[0];
      srcIdx[n*KNN + kk] = g0 + bj;
      d2[bj] = 1e30f;
    }
    __syncthreads();
  }
  if (tid < KNN) {
    const int s = srcIdx[n*KNN + tid];
    emask[n*KNN + tid] = (mask[n] && mask[s]) ? 1.f : 0.f;
  }
}

// ------------------------------ edge features (RBF + pos-embed) ------------------------------
__global__ __launch_bounds__(128)
void edge_feat_kernel(const float* __restrict__ bb5, const int* __restrict__ srcIdx,
                      const float* __restrict__ emask, f16* __restrict__ efeat)
{
  __shared__ float dist[25];
  const int e = blockIdx.x;
  const int dst = e / KNN;
  const int src = srcIdx[e];
  const int tid = threadIdx.x;
  if (tid < 25) {
    const int i = tid / 5, j = tid % 5;
    const float dx = bb5[(size_t)src*15 + i*3 + 0] - bb5[(size_t)dst*15 + j*3 + 0] + 1e-8f;
    const float dy = bb5[(size_t)src*15 + i*3 + 1] - bb5[(size_t)dst*15 + j*3 + 1] + 1e-8f;
    const float dz = bb5[(size_t)src*15 + i*3 + 2] - bb5[(size_t)dst*15 + j*3 + 2] + 1e-8f;
    dist[tid] = sqrtf(dx*dx + dy*dy + dz*dz);
  }
  __syncthreads();
  const float em = emask[e];
  for (int c = tid; c < 416; c += 128) {
    float v;
    if (c < 400) {
      const int p = c >> 4, r = c & 15;
      const float mu = 2.f + (20.f/15.f) * (float)r;   // linspace(2,22,16)
      const float u = (dist[p] - mu) * (1.f/1.25f);    // sigma = 20/16
      v = expf(-u*u);
    } else {
      const int h = c - 400;
      const int k = h & 7;
      const float fr  = expf(-(float)(2*k) * (logf(10000.f)/16.f));
      const float ang = (float)(dst - src) * fr;
      v = (h < 8) ? cosf(ang) : sinf(ang);
    }
    efeat[(size_t)e*416 + c] = (f16)(v * em);
  }
}

// ------------------------------ node-embed input [time(128)|dih(6)|seq(20)|pad] ------------------------------
__global__ void build_nodein_kernel(const float* __restrict__ t, const float* __restrict__ timeW,
                                    const float* __restrict__ dih, const float* __restrict__ seqp,
                                    f16* __restrict__ ni)
{
  const int i = blockIdx.x * blockDim.x + threadIdx.x;
  if (i >= NRES * 160) return;
  const int n = i / 160, c = i % 160;
  const int b = n >> 9;
  float v = 0.f;
  const float twopi = 6.283185307179586f;
  if (c < 64)       v = sinf(twopi * t[b] * timeW[c]);
  else if (c < 128) v = cosf(twopi * t[b] * timeW[c-64]);
  else if (c < 134) v = dih[n*6 + (c-128)];
  else if (c < 154) v = seqp[n*20 + (c-134)];
  ni[i] = (f16)v;
}

// ------------------------------ per-layer point projection ------------------------------
__global__ __launch_bounds__(32)
void points_kernel(const float* __restrict__ s, const float* __restrict__ Wp,
                   const float* __restrict__ bp, const float* __restrict__ R,
                   const float* __restrict__ rigids, float* __restrict__ pg)
{
  __shared__ float pts[12];
  const int n = blockIdx.x, tid = threadIdx.x;
  if (tid < 12) {
    float acc = bp[tid];
    const float* sr = s + (size_t)n*128;
    for (int k = 0; k < 128; ++k) acc += sr[k] * Wp[k*12 + tid];
    pts[tid] = acc;
  }
  __syncthreads();
  if (tid < 12) {
    const int p = tid / 3, i = tid % 3;
    const float* r = R + (size_t)n*9;
    float acc = rigids[(size_t)n*7 + 4 + i];
    for (int j = 0; j < 3; ++j) acc += r[i*3 + j] * pts[p*3 + j];
    pg[(size_t)n*12 + tid] = acc;
  }
}

// ------------------------------ message-input gather [s_src|s_dst|z|d16|rel12|pad] ------------------------------
__global__ __launch_bounds__(128)
void build_min_kernel(const float* __restrict__ s, const float* __restrict__ z,
                      const float* __restrict__ pg, const float* __restrict__ R,
                      const float* __restrict__ rigids, const int* __restrict__ srcIdx,
                      f16* __restrict__ mi)
{
  __shared__ float sh_d[16];
  __shared__ float sh_rel[12];
  const int e = blockIdx.x;
  const int dst = e / KNN;
  const int src = srcIdx[e];
  const int tid = threadIdx.x;
  if (tid < 16) {
    const int i = tid / 4, j = tid % 4;
    const float dx = pg[(size_t)src*12 + i*3+0] - pg[(size_t)dst*12 + j*3+0] + 1e-8f;
    const float dy = pg[(size_t)src*12 + i*3+1] - pg[(size_t)dst*12 + j*3+1] + 1e-8f;
    const float dz = pg[(size_t)src*12 + i*3+2] - pg[(size_t)dst*12 + j*3+2] + 1e-8f;
    sh_d[tid] = sqrtf(dx*dx + dy*dy + dz*dz);
  } else if (tid < 28) {
    const int tt = tid - 16;
    const int p = tt / 3, i = tt % 3;
    float acc = 0.f;
    for (int j = 0; j < 3; ++j)
      acc += R[(size_t)src*9 + j*3 + i] *
             (pg[(size_t)dst*12 + p*3 + j] - rigids[(size_t)src*7 + 4 + j]);
    sh_rel[tt] = acc;
  }
  __syncthreads();
  for (int c = tid; c < 416; c += 128) {
    float v = 0.f;
    if (c < 128)      v = s[(size_t)src*128 + c];
    else if (c < 256) v = s[(size_t)dst*128 + (c-128)];
    else if (c < 384) v = z[(size_t)e*128 + (c-256)];
    else if (c < 400) v = sh_d[c-384];
    else if (c < 412) v = sh_rel[c-400];
    mi[(size_t)e*416 + c] = (f16)v;
  }
}

// ------------------------------ aggregation: dst-major contiguous segment mean ------------------------------
__global__ __launch_bounds__(128)
void agg_kernel(const float* __restrict__ m, const float* __restrict__ emask,
                float* __restrict__ agg)
{
  const int n = blockIdx.x, c = threadIdx.x;
  float s = 0.f, cnt = 0.f;
  for (int k = 0; k < KNN; ++k) {
    s   += m[((size_t)n*KNN + k)*128 + c];
    cnt += emask[n*KNN + k];
  }
  agg[(size_t)n*128 + c] = s / fmaxf(cnt, 1.f);
}

__global__ void build_nu_kernel(const float* __restrict__ s, const float* __restrict__ agg,
                                f16* __restrict__ nu)
{
  const int i = blockIdx.x * blockDim.x + threadIdx.x;
  if (i >= NRES * 256) return;
  const int n = i / 256, c = i % 256;
  const float v = (c < 128) ? s[(size_t)n*128 + c] : agg[(size_t)n*128 + (c-128)];
  nu[i] = (f16)v;
}

__global__ __launch_bounds__(128)
void build_ein_kernel(const float* __restrict__ s, const float* __restrict__ z,
                      const int* __restrict__ srcIdx, f16* __restrict__ ein)
{
  const int e = blockIdx.x;
  const int dst = e / KNN;
  const int src = srcIdx[e];
  for (int c = threadIdx.x; c < 384; c += 128) {
    float v;
    if (c < 128)      v = s[(size_t)src*128 + c];
    else if (c < 256) v = s[(size_t)dst*128 + (c-128)];
    else              v = z[(size_t)e*128 + (c-256)];
    ein[(size_t)e*384 + c] = (f16)v;
  }
}

// ------------------------------ weight conversion f32 -> f16 (zero pad) ------------------------------
__global__ void cvt_w_kernel(const float* __restrict__ W, f16* __restrict__ out,
                             int K, int N, int Kpad, int Npad)
{
  const int i = blockIdx.x * blockDim.x + threadIdx.x;
  if (i >= Kpad * Npad) return;
  const int k = i / Npad, n = i % Npad;
  out[i] = (f16)((k < K && n < N) ? W[k*N + n] : 0.f);
}

__global__ void cvt_bias_kernel(const float* __restrict__ b, float* __restrict__ out,
                                int N, int Npad)
{
  const int i = blockIdx.x * blockDim.x + threadIdx.x;
  if (i < Npad) out[i] = (i < N) ? b[i] : 0.f;
}

__global__ void seq_copy_kernel(const float* __restrict__ sh, float* __restrict__ out)
{
  const int i = blockIdx.x * blockDim.x + threadIdx.x;
  if (i >= NRES * 20) return;
  const int n = i / 20, c = i % 20;
  out[i] = sh[(size_t)n*64 + c];
}

// ------------------------------ host launcher ------------------------------
extern "C" void kernel_launch(void* const* d_in, const int* in_sizes, int n_in,
                              void* d_out, int out_size, void* d_ws, size_t ws_size,
                              hipStream_t stream) {
  (void)in_sizes; (void)n_in; (void)out_size; (void)ws_size;

  // ---- inputs (setup_inputs insertion order; params pytree-flattened, keys sorted) ----
  const float* x       = (const float*)d_in[0];
  const float* bb      = (const float*)d_in[1];
  const float* rigids  = (const float*)d_in[2];
  const float* tvec    = (const float*)d_in[3];
  const float* seqp    = (const float*)d_in[4];
  // params: embed_edge(l0.W,l0.b,l1.W,l1.b,l2.W,l2.b,ln.b,ln.g) = 5..12
  //         embed_node(...) = 13..20 ; layers 21..84 (16 each) ; seq 85,86 ; time_W 87
  const float* eeW0=(const float*)d_in[5],  *eeB0=(const float*)d_in[6];
  const float* eeW1=(const float*)d_in[7],  *eeB1=(const float*)d_in[8];
  const float* eeW2=(const float*)d_in[9],  *eeB2=(const float*)d_in[10];
  const float* eeLnB=(const float*)d_in[11],*eeLnG=(const float*)d_in[12];
  const float* enW0=(const float*)d_in[13], *enB0=(const float*)d_in[14];
  const float* enW1=(const float*)d_in[15], *enB1=(const float*)d_in[16];
  const float* enW2=(const float*)d_in[17], *enB2=(const float*)d_in[18];
  const float* enLnB=(const float*)d_in[19],*enLnG=(const float*)d_in[20];
  const float* seqW=(const float*)d_in[85], *seqB=(const float*)d_in[86];
  const float* timeW=(const float*)d_in[87];
  const unsigned char* mask=(const unsigned char*)d_in[88];

  // ---- workspace bump allocator ----
  char* base = (char*)d_ws; size_t off = 0;
  auto alloc = [&](size_t bytes)->void* {
    void* p = base + off; off = (off + bytes + 255) & ~(size_t)255; return p;
  };
  // f16 weight pool
  f16* wEE0 = (f16*)alloc(416*256*2);
  f16* wEE1 = (f16*)alloc(256*256*2);
  f16* wEE2 = (f16*)alloc(256*128*2);
  f16* wEN0 = (f16*)alloc(160*256*2);
  f16* wEN1 = (f16*)alloc(256*256*2);
  f16* wEN2 = (f16*)alloc(256*128*2);
  f16* wMsg0[4]; f16* wMsg1[4]; f16* wNu[4]; f16* wEd0[4]; f16* wEd1[4];
  for (int l = 0; l < 4; ++l) {
    wMsg0[l]=(f16*)alloc(416*128*2); wMsg1[l]=(f16*)alloc(128*128*2);
    wNu[l]  =(f16*)alloc(256*128*2); wEd0[l]=(f16*)alloc(384*128*2);
    wEd1[l] =(f16*)alloc(128*128*2);
  }
  f16* wSeq = (f16*)alloc(128*64*2);
  float* bSeq = (float*)alloc(64*4);
  // activations / scratch
  float* dih   = (float*)alloc((size_t)NRES*6*4);
  float* bb5   = (float*)alloc((size_t)NRES*15*4);
  float* Rmat  = (float*)alloc((size_t)NRES*9*4);
  float* pg    = (float*)alloc((size_t)NRES*12*4);
  float* nmask = (float*)alloc((size_t)NRES*4);
  int*   srcIdx= (int*)  alloc((size_t)NEDGE*4);
  float* emask = (float*)alloc((size_t)NEDGE*4);
  f16*   ni    = (f16*)  alloc((size_t)NRES*160*2);
  f16*   nh0   = (f16*)  alloc((size_t)NRES*256*2);
  f16*   nh1   = (f16*)  alloc((size_t)NRES*256*2);
  f16*   sh16  = (f16*)  alloc((size_t)NRES*128*2);
  float* spre  = (float*)alloc((size_t)NRES*128*4);
  float* sbuf  = (float*)alloc((size_t)NRES*128*4);
  float* agg   = (float*)alloc((size_t)NRES*128*4);
  float* shout = (float*)alloc((size_t)NRES*64*4);
  f16*   gin   = (f16*)  alloc((size_t)NEDGE*416*2);   // efeat / m_in / e_in / hidden ping
  f16*   ehid  = (f16*)  alloc((size_t)NEDGE*256*2);   // hidden pong
  float* ebuf  = (float*)alloc((size_t)NEDGE*128*4);   // gemm f32 out (m / z-pre)
  float* zbuf  = (float*)alloc((size_t)NEDGE*128*4);   // edge state z

  auto cvtW = [&](const float* W, f16* o, int K, int N, int Kp, int Np){
    int tot = Kp*Np;
    cvt_w_kernel<<<(tot+255)/256, 256, 0, stream>>>(W, o, K, N, Kp, Np);
  };
  auto gemm = [&](const f16* A, const f16* B, const float* bias, float* Cf, f16* Ch,
                  const float* rmask, int M, int N, int K, int relu){
    dim3 grid(M/64, N/64);
    wmma_gemm_kernel<<<grid, 128, 0, stream>>>(A, B, bias, Cf, Ch, rmask, M, N, K, relu);
  };
  auto ln = [&](float* out, const float* in, const float* res, const float* g,
                const float* b, const float* rmask, int rows){
    ln128_kernel<<<rows/4, 128, 0, stream>>>(out, in, res, g, b, rmask, rows);
  };

  // ---- weight conversion ----
  cvtW(eeW0, wEE0, 416,256,416,256); cvtW(eeW1, wEE1, 256,256,256,256);
  cvtW(eeW2, wEE2, 256,128,256,128);
  cvtW(enW0, wEN0, 154,256,160,256); cvtW(enW1, wEN1, 256,256,256,256);
  cvtW(enW2, wEN2, 256,128,256,128);
  for (int l = 0; l < 4; ++l) {
    const int bI = 21 + 16*l;
    cvtW((const float*)d_in[bI+6],  wMsg0[l], 412,128,416,128);
    cvtW((const float*)d_in[bI+8],  wMsg1[l], 128,128,128,128);
    cvtW((const float*)d_in[bI+12], wNu[l],   256,128,256,128);
    cvtW((const float*)d_in[bI+0],  wEd0[l],  384,128,384,128);
    cvtW((const float*)d_in[bI+2],  wEd1[l],  128,128,128,128);
  }
  cvtW(seqW, wSeq, 128,20,128,64);
  cvt_bias_kernel<<<1, 64, 0, stream>>>(seqB, bSeq, 20, 64);

  // ---- geometry / graph ----
  dihedral_kernel<<<(NRES*3+127)/128, 128, 0, stream>>>(bb, mask, dih);
  bb5_kernel<<<(NRES+127)/128, 128, 0, stream>>>(bb, bb5);
  quat_kernel<<<(NRES+127)/128, 128, 0, stream>>>(rigids, Rmat);
  nodemask_kernel<<<(NRES+127)/128, 128, 0, stream>>>(mask, nmask);
  knn_kernel<<<NRES, 128, 0, stream>>>(x, mask, srcIdx, emask);
  edge_feat_kernel<<<NEDGE, 128, 0, stream>>>(bb5, srcIdx, emask, gin);

  // ---- node embedding MLP3 ----
  build_nodein_kernel<<<(NRES*160+255)/256, 256, 0, stream>>>(tvec, timeW, dih, seqp, ni);
  gemm(ni,  wEN0, enB0, nullptr, nh0, nullptr, NRES, 256, 160, 1);
  gemm(nh0, wEN1, enB1, nullptr, nh1, nullptr, NRES, 256, 256, 1);
  gemm(nh1, wEN2, enB2, spre, nullptr, nullptr, NRES, 128, 256, 0);
  ln(sbuf, spre, nullptr, enLnG, enLnB, nullptr, NRES);

  // ---- edge embedding MLP3 ----
  gemm(gin,  wEE0, eeB0, nullptr, ehid, nullptr, NEDGE, 256, 416, 1);
  gemm(ehid, wEE1, eeB1, nullptr, gin,  nullptr, NEDGE, 256, 256, 1); // gin reused as hidden
  gemm(gin,  wEE2, eeB2, ebuf, nullptr, nullptr, NEDGE, 128, 256, 0);
  ln(zbuf, ebuf, nullptr, eeLnG, eeLnB, nullptr, NEDGE);

  // ---- 4 IPMP layers ----
  for (int l = 0; l < 4; ++l) {
    const int bI = 21 + 16*l;
    const float* pW   =(const float*)d_in[bI+14], *pB  =(const float*)d_in[bI+15];
    const float* mB0  =(const float*)d_in[bI+7],  *mB1 =(const float*)d_in[bI+9];
    const float* nuB  =(const float*)d_in[bI+13];
    const float* nLnB =(const float*)d_in[bI+10], *nLnG=(const float*)d_in[bI+11];
    const float* eB0  =(const float*)d_in[bI+1],  *eB1 =(const float*)d_in[bI+3];
    const float* eLnB =(const float*)d_in[bI+4],  *eLnG=(const float*)d_in[bI+5];

    points_kernel<<<NRES, 32, 0, stream>>>(sbuf, pW, pB, Rmat, rigids, pg);
    build_min_kernel<<<NEDGE, 128, 0, stream>>>(sbuf, zbuf, pg, Rmat, rigids, srcIdx, gin);
    gemm(gin,  wMsg0[l], mB0, nullptr, ehid, nullptr, NEDGE, 128, 416, 1);
    gemm(ehid, wMsg1[l], mB1, ebuf, nullptr, emask, NEDGE, 128, 128, 0);
    agg_kernel<<<NRES, 128, 0, stream>>>(ebuf, emask, agg);
    build_nu_kernel<<<(NRES*256+255)/256, 256, 0, stream>>>(sbuf, agg, nh0);
    gemm(nh0, wNu[l], nuB, spre, nullptr, nullptr, NRES, 128, 256, 0);
    ln(sbuf, spre, sbuf, nLnG, nLnB, nmask, NRES);       // s = LN(s + upd) * mask
    build_ein_kernel<<<NEDGE, 128, 0, stream>>>(sbuf, zbuf, srcIdx, gin);
    gemm(gin,  wEd0[l], eB0, nullptr, ehid, nullptr, NEDGE, 128, 384, 1);
    gemm(ehid, wEd1[l], eB1, ebuf, nullptr, nullptr, NEDGE, 128, 128, 0);
    ln(zbuf, ebuf, zbuf, eLnG, eLnB, emask, NEDGE);      // z = LN(z + upd) * emask
  }

  // ---- sequence head ----
  cvt_a_kernel<<<(NRES*128+255)/256, 256, 0, stream>>>(sbuf, sh16, NRES*128);
  gemm(sh16, wSeq, bSeq, shout, nullptr, nullptr, NRES, 64, 128, 0);
  seq_copy_kernel<<<(NRES*20+255)/256, 256, 0, stream>>>(shout, (float*)d_out);
}
